// MoELayerParallel_33990371180786
// MI455X (gfx1250) — matrix operand order repaired and checked by reading, not compiled, not run on hardware
//
#include <hip/hip_runtime.h>

typedef __attribute__((ext_vector_type(16))) __bf16 v16bf;
typedef __attribute__((ext_vector_type(8)))  __bf16 v8bf;
typedef __attribute__((ext_vector_type(8)))  float  v8f;

// Problem dims
constexpr int Bq = 4, Sq = 2048, Dq = 768, Hq = 3072, Eq = 8;
constexpr int Nq = Bq * Sq;              // 8192 tokens
constexpr float LB_SCALE = 0.01f;

// Tiling: BM x BN block tile, BK k-step, 256 threads = 8 waves (wave32)
constexpr int BM = 64, BN = 64, BK = 32;

// Workspace layout (bytes)
constexpr size_t OFF_COUNTS = 0;                                   // E ints
constexpr size_t OFF_SUMS   = 256;                                 // E floats
constexpr size_t OFF_LIST   = 512;                                 // E*N ints
constexpr size_t OFF_GVALS  = OFF_LIST  + (size_t)Eq * Nq * 4;     // E*N floats
constexpr size_t OFF_XBF    = OFF_GVALS + (size_t)Eq * Nq * 4;     // N*D bf16
constexpr size_t OFF_HBUF   = OFF_XBF   + (size_t)Nq * Dq * 2;     // N*2*H bf16

__device__ inline v8f v8f_zero() {
  v8f z;
#pragma unroll
  for (int i = 0; i < 8; ++i) z[i] = 0.0f;
  return z;
}
__device__ inline v8bf v8bf_zero() {
  v8bf z;
#pragma unroll
  for (int i = 0; i < 8; ++i) z[i] = (__bf16)0.0f;
  return z;
}

// ---------------------------------------------------------------------------
// Stage one 8-element half of a B fragment (fp32 -> bf16), fragment-major LDS.
// B 32x16 bf16 layout: fraglane fl -> N = fl&15, elems 0..7 K = kh8..kh8+7, elems 8..15 K = 16+kh8.., kh8 = 8*(fl>=16).
// 256 threads cover one full 32x64 B tile (4 ntiles x 32 lanes x 2 halves).
// ---------------------------------------------------------------------------
__device__ inline void stage_b_half(v16bf* BF, const float* __restrict__ wsrc,
                                    int ldw, int n0, int k0, int tid) {
  int fl    = tid & 31;
  int ntile = (tid >> 5) & 3;
  int kh    = (tid >> 7) & 1;
  int n     = n0 + ntile * 16 + (fl & 15);
  int kbase = k0 + ((fl & 16) >> 1) + kh * 16;
  const float* src = wsrc + (size_t)kbase * ldw + n;
  __builtin_prefetch(src + (size_t)BK * ldw, 0, 1);   // next k-chunk
  v8bf v;
#pragma unroll
  for (int j = 0; j < 8; ++j) v[j] = (__bf16)src[(size_t)j * ldw];
  ((v8bf*)BF)[((ntile * 32 + fl) << 1) | kh] = v;     // one ds_store_b128
}

// ---------------------------------------------------------------------------
// Kernel 0: zero y / counters, convert x -> bf16 once
// ---------------------------------------------------------------------------
__global__ void moe_init(const float* __restrict__ x, __bf16* __restrict__ xbf,
                         float* __restrict__ y, int* __restrict__ counts,
                         float* __restrict__ sums) {
  int i = blockIdx.x * blockDim.x + threadIdx.x;
  int stride = gridDim.x * blockDim.x;
  for (int idx = i; idx < Nq * Dq; idx += stride) {
    xbf[idx] = (__bf16)x[idx];
    y[idx] = 0.0f;
  }
  if (i < Eq) { counts[i] = 0; sums[i] = 0.0f; }
}

// ---------------------------------------------------------------------------
// Kernel 1: gating — one wave32 per token
// ---------------------------------------------------------------------------
__global__ __launch_bounds__(256)
void moe_gate(const float* __restrict__ x, const float* __restrict__ gate_w,
              const float* __restrict__ noise_w, const float* __restrict__ noise,
              int* __restrict__ counts, float* __restrict__ sums,
              int* __restrict__ list, float* __restrict__ gvals) {
  __shared__ float gws[Eq * Dq];  // 24 KB
  for (int i = threadIdx.x; i < Eq * Dq; i += blockDim.x) gws[i] = gate_w[i];
  __syncthreads();

  int wave = threadIdx.x >> 5, lane = threadIdx.x & 31;
  int t = blockIdx.x * 8 + wave;
  if (t >= Nq) return;

  float acc[Eq];
#pragma unroll
  for (int e = 0; e < Eq; ++e) acc[e] = 0.0f;
  for (int d = lane; d < Dq; d += 32) {
    float xv = x[(size_t)t * Dq + d];
#pragma unroll
    for (int e = 0; e < Eq; ++e) acc[e] += xv * gws[e * Dq + d];
  }
#pragma unroll
  for (int e = 0; e < Eq; ++e)
    for (int off = 16; off > 0; off >>= 1) acc[e] += __shfl_xor(acc[e], off, 32);

  if (lane == 0) {
    float m = acc[0];
#pragma unroll
    for (int e = 1; e < Eq; ++e) m = fmaxf(m, acc[e]);
    float p[Eq], den = 0.0f;
#pragma unroll
    for (int e = 0; e < Eq; ++e) { p[e] = __expf(acc[e] - m); den += p[e]; }
    float inv = 1.0f / den;
#pragma unroll
    for (int e = 0; e < Eq; ++e) atomicAdd(&sums[e], p[e] * inv);

    float ln[Eq];
#pragma unroll
    for (int e = 0; e < Eq; ++e) ln[e] = acc[e] + noise[(size_t)t * Eq + e] * noise_w[e];

    int e0 = 0;
#pragma unroll
    for (int e = 1; e < Eq; ++e) if (ln[e] > ln[e0]) e0 = e;
    int e1 = (e0 == 0) ? 1 : 0;
#pragma unroll
    for (int e = 0; e < Eq; ++e) if (e != e0 && ln[e] > ln[e1]) e1 = e;

    float g0 = 1.0f / (1.0f + __expf(ln[e1] - ln[e0]));
    float g1 = 1.0f - g0;

    int s0 = atomicAdd(&counts[e0], 1);
    list[e0 * Nq + s0]  = (t << 1);
    gvals[e0 * Nq + s0] = g0;
    int s1 = atomicAdd(&counts[e1], 1);
    list[e1 * Nq + s1]  = (t << 1) | 1;
    gvals[e1 * Nq + s1] = g1;
  }
}

// ---------------------------------------------------------------------------
// Kernel 2: pass1 — h = (Xg @ w1 + b1) * silu(Xg @ w2 + b2), stored bf16
// grid: (H/BN, N/BM, E), block 256 (8 waves)
// ---------------------------------------------------------------------------
__global__ __launch_bounds__(256)
void moe_pass1(const __bf16* __restrict__ xbf,
               const float* __restrict__ w1, const float* __restrict__ b1,
               const float* __restrict__ w2, const float* __restrict__ b2,
               const int* __restrict__ counts, const int* __restrict__ list,
               __bf16* __restrict__ hbuf) {
  int e = blockIdx.z;
  int cnt = counts[e];
  int m0 = blockIdx.y * BM;
  if (m0 >= cnt) return;
  int n0 = blockIdx.x * BN;
  bool full = (m0 + BM) <= cnt;          // block-uniform

  __shared__ v16bf AsF[128];             // 4 mtiles x 32 lanes, fragment-major
  __shared__ v16bf B1F[128];             // 4 ntiles x 32 lanes
  __shared__ v16bf B2F[128];
  __shared__ int rowinfo[BM];

  int tid = threadIdx.x, wave = tid >> 5, lane = tid & 31;
  if (tid < BM) rowinfo[tid] = (m0 + tid < cnt) ? list[e * Nq + m0 + tid] : -1;
  __syncthreads();

  // A-staging identity: waves 0-3 move the gathered A tile (bf16, no convert).
  // A 16x32 bf16 frag: lane L -> M = L&15; elems 0..7 = K kh..kh+7,
  // elems 8..15 = K 16+kh..16+kh+7, kh = 8*(L>=16).
  int amtile = wave & 3;
  int arow   = amtile * 16 + (lane & 15);
  int ainfo  = rowinfo[arow];
  const __bf16* agsrc =
      xbf + (size_t)((ainfo >= 0 ? ainfo : 0) >> 1) * Dq + ((lane & 16) >> 1);
  unsigned alds = (unsigned)(uintptr_t)&AsF[0] + (unsigned)((amtile * 32 + lane) * 32);

  const float* w1e = w1 + (size_t)e * Dq * Hq;
  const float* w2e = w2 + (size_t)e * Dq * Hq;

  v8f acc1[2] = {v8f_zero(), v8f_zero()};
  v8f acc2[2] = {v8f_zero(), v8f_zero()};

  for (int k0 = 0; k0 < Dq; k0 += BK) {
    __syncthreads();                     // previous fragments consumed
    if (tid < 128) {
      if (full) {
        // CDNA5 async DMA: per-lane 16B global -> LDS, tracked by ASYNCcnt.
        // offset:16 adds 16 to BOTH lds and global addr; bias global by +16
        // so elems 8..15 land at lds+16 from global+32 (K jump of 16).
        unsigned long long g0 = (unsigned long long)(uintptr_t)(agsrc + k0);
        asm volatile(
            "global_load_async_to_lds_b128 %0, %1, off\n\t"
            "global_load_async_to_lds_b128 %0, %2, off offset:16"
            :: "v"(alds), "v"(g0), "v"(g0 + 16) : "memory");
      } else {
        v8bf c0 = v8bf_zero(), c1 = c0;
        if (ainfo >= 0) {
          const v8bf* s = (const v8bf*)(agsrc + k0);
          c0 = s[0];                     // K kh..kh+7
          c1 = s[2];                     // K 16+kh..16+kh+7 (+32B)
        }
        v8bf* dst = (v8bf*)AsF + (amtile * 32 + lane) * 2;
        dst[0] = c0;
        dst[1] = c1;
      }
    }
    stage_b_half(B1F, w1e, Hq, n0, k0, tid);
    stage_b_half(B2F, w2e, Hq, n0, k0, tid);
    asm volatile("s_wait_asynccnt 0" ::: "memory");
    __syncthreads();

    v16bf fb1 = B1F[(wave & 3) * 32 + lane];   // 2x ds_load_b128
    v16bf fb2 = B2F[(wave & 3) * 32 + lane];
    int mbase = (wave >> 2) * 2;
#pragma unroll
    for (int mt = 0; mt < 2; ++mt) {
      v16bf fa = AsF[(mbase + mt) * 32 + lane];
      acc1[mt] = __builtin_amdgcn_wmma_f32_16x16x32_bf16(false, fa, false, fb1,
                                                         (short)0, acc1[mt], false, false);
      acc2[mt] = __builtin_amdgcn_wmma_f32_16x16x32_bf16(false, fa, false, fb2,
                                                         (short)0, acc2[mt], false, false);
    }
  }

  // Epilogue: bias + SwiGLU, store bf16 rows indexed by assignment (t*2+s)
  int wn = (wave & 3) * 16, wm = (wave >> 2) * 32;
  int nglob = n0 + wn + (lane & 15);
  float bias1 = b1[e * Hq + nglob];
  float bias2 = b2[e * Hq + nglob];
#pragma unroll
  for (int mt = 0; mt < 2; ++mt) {
#pragma unroll
    for (int r = 0; r < 8; ++r) {
      int mrow = wm + mt * 16 + r + ((lane & 16) >> 1);
      int info = rowinfo[mrow];
      if (info >= 0) {
        float c1 = acc1[mt][r] + bias1;
        float c2 = acc2[mt][r] + bias2;
        float h = c1 * (c2 / (1.0f + __expf(-c2)));   // c1 * silu(c2)
        hbuf[(size_t)info * Hq + nglob] = (__bf16)h;
      }
    }
  }
}

// ---------------------------------------------------------------------------
// Kernel 3: pass2 — y[t] += gate * (h @ wp + bp)
// grid: (D/BN, N/BM, E), block 256
// ---------------------------------------------------------------------------
__global__ __launch_bounds__(256)
void moe_pass2(const __bf16* __restrict__ hbuf,
               const float* __restrict__ wp, const float* __restrict__ bp,
               const int* __restrict__ counts, const int* __restrict__ list,
               const float* __restrict__ gvals, float* __restrict__ y) {
  int e = blockIdx.z;
  int cnt = counts[e];
  int m0 = blockIdx.y * BM;
  if (m0 >= cnt) return;
  int n0 = blockIdx.x * BN;
  bool full = (m0 + BM) <= cnt;

  __shared__ v16bf AsF[128];
  __shared__ v16bf BpF[128];
  __shared__ int rowinfo[BM];
  __shared__ float rowg[BM];

  int tid = threadIdx.x, wave = tid >> 5, lane = tid & 31;
  if (tid < BM) {
    int r = m0 + tid;
    bool ok = (r < cnt);
    rowinfo[tid] = ok ? list[e * Nq + r] : -1;
    rowg[tid]    = ok ? gvals[e * Nq + r] : 0.0f;
  }
  __syncthreads();

  int amtile = wave & 3;
  int arow   = amtile * 16 + (lane & 15);
  int ainfo  = rowinfo[arow];
  const __bf16* agsrc =
      hbuf + (size_t)(ainfo >= 0 ? ainfo : 0) * Hq + ((lane & 16) >> 1);
  unsigned alds = (unsigned)(uintptr_t)&AsF[0] + (unsigned)((amtile * 32 + lane) * 32);

  const float* wpe = wp + (size_t)e * Hq * Dq;

  v8f acc[2] = {v8f_zero(), v8f_zero()};

  for (int k0 = 0; k0 < Hq; k0 += BK) {
    __syncthreads();
    if (tid < 128) {
      if (full) {
        unsigned long long g0 = (unsigned long long)(uintptr_t)(agsrc + k0);
        asm volatile(
            "global_load_async_to_lds_b128 %0, %1, off\n\t"
            "global_load_async_to_lds_b128 %0, %2, off offset:16"
            :: "v"(alds), "v"(g0), "v"(g0 + 16) : "memory");
      } else {
        v8bf c0 = v8bf_zero(), c1 = c0;
        if (ainfo >= 0) {
          const v8bf* s = (const v8bf*)(agsrc + k0);
          c0 = s[0];
          c1 = s[2];
        }
        v8bf* dst = (v8bf*)AsF + (amtile * 32 + lane) * 2;
        dst[0] = c0;
        dst[1] = c1;
      }
    }
    stage_b_half(BpF, wpe, Dq, n0, k0, tid);
    asm volatile("s_wait_asynccnt 0" ::: "memory");
    __syncthreads();

    v16bf fb = BpF[(wave & 3) * 32 + lane];
    int mbase = (wave >> 2) * 2;
#pragma unroll
    for (int mt = 0; mt < 2; ++mt) {
      v16bf fa = AsF[(mbase + mt) * 32 + lane];
      acc[mt] = __builtin_amdgcn_wmma_f32_16x16x32_bf16(false, fa, false, fb,
                                                        (short)0, acc[mt], false, false);
    }
  }

  int wn = (wave & 3) * 16, wm = (wave >> 2) * 32;
  int nglob = n0 + wn + (lane & 15);
  float bias = bp[e * Dq + nglob];
#pragma unroll
  for (int mt = 0; mt < 2; ++mt) {
#pragma unroll
    for (int r = 0; r < 8; ++r) {
      int mrow = wm + mt * 16 + r + ((lane & 16) >> 1);
      int info = rowinfo[mrow];
      if (info >= 0) {
        int t = info >> 1;
        float v = rowg[mrow] * (acc[mt][r] + bias);
        atomicAdd(&y[(size_t)t * Dq + nglob], v);
      }
    }
  }
}

// ---------------------------------------------------------------------------
// Kernel 4: load-balance loss scalar
// ---------------------------------------------------------------------------
__global__ void moe_loss(const float* __restrict__ sums, float* __restrict__ out) {
  if (threadIdx.x == 0 && blockIdx.x == 0) {
    float acc = 0.0f;
    for (int e = 0; e < Eq; ++e) {
      float d = sums[e] / (float)Nq - 1.0f / (float)Eq;
      acc += d * d;
    }
    out[0] = (acc / (float)Eq) * LB_SCALE;
  }
}

// ---------------------------------------------------------------------------
extern "C" void kernel_launch(void* const* d_in, const int* in_sizes, int n_in,
                              void* d_out, int out_size, void* d_ws, size_t ws_size,
                              hipStream_t stream) {
  const float* x   = (const float*)d_in[0];
  const float* gw  = (const float*)d_in[1];
  const float* nw  = (const float*)d_in[2];
  const float* noi = (const float*)d_in[3];
  const float* w1  = (const float*)d_in[4];
  const float* b1  = (const float*)d_in[5];
  const float* w2  = (const float*)d_in[6];
  const float* b2  = (const float*)d_in[7];
  const float* wp  = (const float*)d_in[8];
  const float* bp  = (const float*)d_in[9];

  float* y = (float*)d_out;
  char* ws = (char*)d_ws;
  int*    counts = (int*)(ws + OFF_COUNTS);
  float*  sums   = (float*)(ws + OFF_SUMS);
  int*    list   = (int*)(ws + OFF_LIST);
  float*  gvals  = (float*)(ws + OFF_GVALS);
  __bf16* xbf    = (__bf16*)(ws + OFF_XBF);
  __bf16* hbuf   = (__bf16*)(ws + OFF_HBUF);

  moe_init<<<4096, 256, 0, stream>>>(x, xbf, y, counts, sums);
  moe_gate<<<Nq / 8, 256, 0, stream>>>(x, gw, nw, noi, counts, sums, list, gvals);
  moe_pass1<<<dim3(Hq / BN, Nq / BM, Eq), 256, 0, stream>>>(xbf, w1, b1, w2, b2,
                                                            counts, list, hbuf);
  moe_pass2<<<dim3(Dq / BN, Nq / BM, Eq), 256, 0, stream>>>(hbuf, wp, bp, counts,
                                                            list, gvals, y);
  moe_loss<<<1, 32, 0, stream>>>(sums, y + (size_t)Nq * Dq);
}
